// AttentionPooling_56100862820558
// MI455X (gfx1250) — matrix operand order repaired and checked by reading, not compiled, not run on hardware
//
#include <hip/hip_runtime.h>
#include <math.h>

#define H          256
#define H2         128
#define NUM_GRAPHS 4096
#define LDS_K      (H + 8)   // padded k-stride: 528B row pitch -> conflict-free B gathers

typedef __attribute__((ext_vector_type(16))) __bf16 v16bf;
typedef __attribute__((ext_vector_type(8)))  float  v8f;

static __device__ __forceinline__ unsigned short f32_to_bf16_bits(float f) {
    unsigned u = __builtin_bit_cast(unsigned, f);
    unsigned r = u + 0x7FFFu + ((u >> 16) & 1u);   // round-to-nearest-even
    return (unsigned short)(r >> 16);
}
static __device__ __forceinline__ __bf16 bits_to_bf16(unsigned short s) {
    return __builtin_bit_cast(__bf16, s);
}
// monotone order-preserving f32 <-> u32 map for atomicMax on floats
static __device__ __forceinline__ unsigned f32_map(float f) {
    unsigned u = __builtin_bit_cast(unsigned, f);
    return (u & 0x80000000u) ? ~u : (u | 0x80000000u);
}
static __device__ __forceinline__ float f32_unmap(unsigned m) {
    unsigned u = (m & 0x80000000u) ? (m ^ 0x80000000u) : ~m;
    return __builtin_bit_cast(float, u);
}

// ---------------------------------------------------------------------------
// Pass 0: zero segment tables + pooled output
// ---------------------------------------------------------------------------
__global__ void ap_init_kernel(unsigned* __restrict__ segmax_u,
                               float* __restrict__ segsum,
                               float* __restrict__ pooled) {
    int i = blockIdx.x * blockDim.x + threadIdx.x;
    if (i < NUM_GRAPHS) { segmax_u[i] = 0u; segsum[i] = 0.f; }
    if (i < NUM_GRAPHS * H) pooled[i] = 0.f;
}

// ---------------------------------------------------------------------------
// Pass 1: gate MLP scores via bf16 WMMA.  Block = 256 thr (8 waves), 128 nodes.
// Each wave: 16-node strip, h(16x128) in 8 f32 accumulators via
// v_wmma_f32_16x16x32_bf16, then tanh + W2 dot + shfl reduction.
// ---------------------------------------------------------------------------
__global__ __launch_bounds__(256)
void ap_gate_scores_kernel(const float* __restrict__ x,
                           const float* __restrict__ W1,   // (256,128) row-major
                           const float* __restrict__ b1,   // (128,)
                           const float* __restrict__ W2,   // (128,)
                           const float* __restrict__ b2,   // (1,)
                           float* __restrict__ scores,     // (N,)
                           int N)
{
    __shared__ unsigned short w1s[H2 * LDS_K];   // bf16 W1, transposed [n][k], padded

    const int tid = threadIdx.x;
    for (int i = tid; i < H * H2; i += 256) {    // coalesced global read of W1[k][n]
        int k = i >> 7, n = i & 127;
        w1s[n * LDS_K + k] = f32_to_bf16_bits(W1[i]);
    }
    __syncthreads();

    const int wave = tid >> 5;
    const int lane = tid & 31;
    const int hlf  = lane >> 4;     // half-wave: 0 = lanes 0-15, 1 = lanes 16-31
    const int lm   = lane & 15;
    const int nodeBase = blockIdx.x * 128 + wave * 16;

    v8f acc[8];
    #pragma unroll
    for (int t = 0; t < 8; ++t) acc[t] = (v8f){0,0,0,0,0,0,0,0};

    int arow = nodeBase + lm;                    // A-matrix row for this lane (clamped tail)
    if (arow > N - 1) arow = N - 1;
    const float* xrow = x + (long)arow * H;

    for (int kc = 0; kc < H; kc += 32) {
        // A fragment 16x32 bf16: half 0 -> K 0-15, half 1 -> K 16-31 (contiguous, matches B)
        v16bf a;
        const int klo = kc + hlf * 16;
        const int khi = kc + hlf * 16 + 8;
        #pragma unroll
        for (int e = 0; e < 8; ++e) {
            a[e]     = bits_to_bf16(f32_to_bf16_bits(xrow[klo + e]));
            a[8 + e] = bits_to_bf16(f32_to_bf16_bits(xrow[khi + e]));
        }
        // 8 column tiles of W1: B fragment 32x16, lane = N column, pair-packed K
        #pragma unroll
        for (int t = 0; t < 8; ++t) {
            const int n = t * 16 + lm;
            const unsigned short* p = &w1s[n * LDS_K + kc + hlf * 16];
            v16bf b;
            #pragma unroll
            for (int j = 0; j < 16; ++j) b[j] = bits_to_bf16(p[j]);
            acc[t] = __builtin_amdgcn_wmma_f32_16x16x32_bf16(
                         false, a, false, b, (short)0, acc[t], false, false);
        }
    }

    // h = tanh(acc + b1); per-lane partial score = sum over this lane's N columns
    float spart[8];
    #pragma unroll
    for (int r = 0; r < 8; ++r) spart[r] = 0.f;
    #pragma unroll
    for (int t = 0; t < 8; ++t) {
        const int n  = t * 16 + lm;
        const float bb = b1[n];
        const float ww = W2[n];
        #pragma unroll
        for (int r = 0; r < 8; ++r) {
            float hv = tanhf(acc[t][r] + bb);
            spart[r] += hv * ww;
        }
    }
    // reduce over the 16 lanes of each half-wave (C layout: VGPR r = row r / r+8)
    #pragma unroll
    for (int m = 8; m >= 1; m >>= 1) {
        #pragma unroll
        for (int r = 0; r < 8; ++r)
            spart[r] += __shfl_xor(spart[r], m, 32);
    }
    if (lm == 0) {
        const float bb2 = b2[0];
        #pragma unroll
        for (int r = 0; r < 8; ++r) {
            int node = nodeBase + hlf * 8 + r;
            if (node < N) scores[node] = spart[r] + bb2;
        }
    }
}

// ---------------------------------------------------------------------------
// Pass 2: segment max (atomicMax on order-preserving uint map; 0 == "empty")
// ---------------------------------------------------------------------------
__global__ void ap_segmax_kernel(const float* __restrict__ scores,
                                 const long long* __restrict__ batch,
                                 unsigned* __restrict__ segmax_u, int N) {
    int i = blockIdx.x * blockDim.x + threadIdx.x;
    if (i >= N) return;
    atomicMax(&segmax_u[(int)batch[i]], f32_map(scores[i]));
}

// ---------------------------------------------------------------------------
// Pass 3: ex = exp(s - segmax), segment sum
// ---------------------------------------------------------------------------
__global__ void ap_expsum_kernel(const float* __restrict__ scores,
                                 const long long* __restrict__ batch,
                                 const unsigned* __restrict__ segmax_u,
                                 float* __restrict__ ex,
                                 float* __restrict__ segsum, int N) {
    int i = blockIdx.x * blockDim.x + threadIdx.x;
    if (i >= N) return;
    int g = (int)batch[i];
    unsigned mu = segmax_u[g];
    float m = (mu == 0u) ? 0.f : f32_unmap(mu);
    float e = expf(scores[i] - m);
    ex[i] = e;
    atomicAdd(&segsum[g], e);
}

// ---------------------------------------------------------------------------
// Pass 4: weights = ex / segsum[batch]
// ---------------------------------------------------------------------------
__global__ void ap_weights_kernel(const float* __restrict__ ex,
                                  const long long* __restrict__ batch,
                                  const float* __restrict__ segsum,
                                  float* __restrict__ weights, int N) {
    int i = blockIdx.x * blockDim.x + threadIdx.x;
    if (i >= N) return;
    weights[i] = ex[i] / segsum[(int)batch[i]];
}

// ---------------------------------------------------------------------------
// Pass 5: weighted pooling. 1 thread per H column, 128 sorted nodes per block;
// register accumulation, atomic flush only at segment boundaries.
// ---------------------------------------------------------------------------
__global__ __launch_bounds__(256)
void ap_pool_kernel(const float* __restrict__ x,
                    const long long* __restrict__ batch,
                    const float* __restrict__ weights,
                    float* __restrict__ pooled, int N)
{
    __shared__ int   seg[128];
    __shared__ float wgt[128];
    const int base = blockIdx.x * 128;
    const int c    = threadIdx.x;           // column 0..255
    if (c < 128) {
        int idx = base + c;
        if (idx < N) { seg[c] = (int)batch[idx]; wgt[c] = weights[idx]; }
        else         { seg[c] = -1;              wgt[c] = 0.f; }
    }
    __syncthreads();

    int cnt = N - base; if (cnt > 128) cnt = 128;
    if (cnt <= 0) return;

    float acc = 0.f;
    int   cur = seg[0];
    for (int i = 0; i < cnt; ++i) {
        int g = seg[i];
        if (g != cur) {
            atomicAdd(&pooled[cur * H + c], acc);
            acc = 0.f;
            cur = g;
        }
        acc += wgt[i] * x[(long)(base + i) * H + c];
    }
    atomicAdd(&pooled[cur * H + c], acc);
}

// ---------------------------------------------------------------------------
extern "C" void kernel_launch(void* const* d_in, const int* in_sizes, int n_in,
                              void* d_out, int out_size, void* d_ws, size_t ws_size,
                              hipStream_t stream)
{
    (void)n_in; (void)out_size; (void)ws_size;
    const float*     x     = (const float*)d_in[0];
    const long long* batch = (const long long*)d_in[1];
    const float*     W1    = (const float*)d_in[2];
    const float*     b1    = (const float*)d_in[3];
    const float*     W2    = (const float*)d_in[4];
    const float*     b2    = (const float*)d_in[5];

    const int N = in_sizes[1];   // number of nodes (batch length)

    float* pooled  = (float*)d_out;                 // (4096, 256)
    float* weights = pooled + NUM_GRAPHS * H;       // (N,)

    char* ws = (char*)d_ws;
    float*    scores = (float*)ws;     ws += (size_t)N * sizeof(float);
    float*    ex     = (float*)ws;     ws += (size_t)N * sizeof(float);
    unsigned* smax   = (unsigned*)ws;  ws += (size_t)NUM_GRAPHS * sizeof(unsigned);
    float*    ssum   = (float*)ws;

    const int nbInit = (NUM_GRAPHS * H + 255) / 256;
    const int nb128  = (N + 127) / 128;
    const int nbN    = (N + 255) / 256;

    ap_init_kernel       <<<nbInit, 256, 0, stream>>>(smax, ssum, pooled);
    ap_gate_scores_kernel<<<nb128, 256, 0, stream>>>(x, W1, b1, W2, b2, scores, N);
    ap_segmax_kernel     <<<nbN,   256, 0, stream>>>(scores, batch, smax, N);
    ap_expsum_kernel     <<<nbN,   256, 0, stream>>>(scores, batch, smax, ex, ssum, N);
    ap_weights_kernel    <<<nbN,   256, 0, stream>>>(ex, batch, ssum, weights, N);
    ap_pool_kernel       <<<nb128, 256, 0, stream>>>(x, batch, weights, pooled, N);
}
